// PriceRegressor_34471407517779
// MI455X (gfx1250) — matrix operand order repaired and checked by reading, not compiled, not run on hardware
//
#include <hip/hip_runtime.h>
#include <hip/hip_bf16.h>

#define BATCH   64
#define SEQL    1024
#define ROWS    (BATCH*SEQL)      // 65536
#define DMODEL  256
#define DINNER  512
#define NST     64
#define DTRANK  16
#define DBC     80                // dt(16) + B(64), C computed only for last token

typedef _Float16 half8  __attribute__((ext_vector_type(8)));
typedef _Float16 half16 __attribute__((ext_vector_type(16)));
typedef float    float8 __attribute__((ext_vector_type(8)));

// ---------------------------------------------------------------------------
// Pack a f32 weight matrix (K x Ncols, leading dim ldw, starting at coloff)
// into this file's B-fragment order (contiguous K, same as the A loads) for v_wmma_f32_16x16x32_f16:
//   [ntile][kchunk][lane][16 halves], lane&15 = column, lane>=16 -> K+16,
//   halves 0..15 = consecutive K.
// ---------------------------------------------------------------------------
__global__ void pack_b_f16(const float* __restrict__ W, int ldw, int coloff,
                           int K, _Float16* __restrict__ out, int total) {
  int idx = blockIdx.x * blockDim.x + threadIdx.x;
  if (idx >= total) return;
  int nk      = K >> 5;
  int perTile = nk * 512;           // 32 lanes * 16 halves per kchunk
  int ntile   = idx / perTile;
  int rem     = idx % perTile;
  int kc      = rem / 512;
  int r2      = rem % 512;
  int lane    = r2 >> 4;
  int h       = r2 & 15;
  int n       = ntile * 16 + (lane & 15);
  int k       = kc * 32 + ((lane >= 16) ? 16 : 0) + h;
  out[idx] = (_Float16)W[(size_t)k * ldw + coloff + n];
}

// ---------------------------------------------------------------------------
// Embed: h = x @ embed_w + embed_b  (K=20, tiny -> VALU). Writes f16 for the
// WMMA GEMM and captures the f32 residual at the last timestep.
// ---------------------------------------------------------------------------
__global__ void embed_kernel(const float* __restrict__ x, const float* __restrict__ w,
                             const float* __restrict__ bias, _Float16* __restrict__ h16,
                             float* __restrict__ resid) {
  int idx = blockIdx.x * blockDim.x + threadIdx.x;
  if (idx >= ROWS * DMODEL) return;
  int c   = idx & (DMODEL - 1);
  int row = idx >> 8;
  const float* xr = x + (size_t)row * 20;
  float acc = bias[c];
  #pragma unroll
  for (int f = 0; f < 20; ++f) acc += xr[f] * w[f * DMODEL + c];
  h16[idx] = (_Float16)acc;
  if ((row & (SEQL - 1)) == SEQL - 1) resid[(row >> 10) * DMODEL + c] = acc;
}

// ---------------------------------------------------------------------------
// Generic WMMA GEMM:  C(MxN) = A(MxK,f16,row-major) @ Bp(packed f16)
// One wave computes a 16 x (16*NT) tile; f32 accumulation via
// v_wmma_f32_16x16x32_f16.
// ---------------------------------------------------------------------------
template <int NT, typename OutT>
__global__ void gemm_wmma(const _Float16* __restrict__ A, const _Float16* __restrict__ Bp,
                          OutT* __restrict__ C, int M, int N, int K, int ldc) {
  const int lane   = threadIdx.x & 31;
  const int wave   = blockIdx.x * (blockDim.x >> 5) + (threadIdx.x >> 5);
  const int wavesN = N / (16 * NT);
  const int mt     = wave / wavesN;
  const int ng     = wave % wavesN;
  if (mt * 16 >= M) return;
  const int nk = K >> 5;

  float8 acc[NT];
  #pragma unroll
  for (int j = 0; j < NT; ++j)
    #pragma unroll
    for (int r = 0; r < 8; ++r) acc[j][r] = 0.f;

  const int row = mt * 16 + (lane & 15);
  // A-fragment: lanes 0-15 K=0..15 ; lanes 16-31 K=16..31
  const _Float16* arow = A + (size_t)row * K + ((lane < 16) ? 0 : 16);

  for (int kc = 0; kc < nk; ++kc) {
    half16 afrag;
    const half8 lo = *(const half8*)(arow + kc * 32);
    const half8 hi = *(const half8*)(arow + kc * 32 + 8);
    #pragma unroll
    for (int i = 0; i < 8; ++i) { afrag[i] = lo[i]; afrag[i + 8] = hi[i]; }
    #pragma unroll
    for (int j = 0; j < NT; ++j) {
      int ntile    = ng * NT + j;
      half16 bfrag = *(const half16*)(Bp + ((size_t)(ntile * nk + kc) * 32 + lane) * 16);
      acc[j] = __builtin_amdgcn_wmma_f32_16x16x32_f16(
          false, afrag, false, bfrag, (short)0, acc[j], false, false);
    }
  }
  // C/D layout: VGPR r -> M = r (lanes 0-15) / M = 8+r (lanes 16-31), N = lane&15
  const int mbase = mt * 16 + ((lane >= 16) ? 8 : 0);
  const int nc    = lane & 15;
  #pragma unroll
  for (int j = 0; j < NT; ++j) {
    int n0 = (ng * NT + j) * 16;
    #pragma unroll
    for (int r = 0; r < 8; ++r)
      C[(size_t)(mbase + r) * ldc + n0 + nc] = (OutT)acc[j][r];
  }
}

// ---------------------------------------------------------------------------
// Depthwise causal conv (width 4) + SiLU.
// ---------------------------------------------------------------------------
__global__ void conv_silu(const _Float16* __restrict__ xi, const float* __restrict__ cw,
                          const float* __restrict__ cb, _Float16* __restrict__ xc) {
  int idx = blockIdx.x * blockDim.x + threadIdx.x;
  if (idx >= ROWS * DINNER) return;
  int d  = idx & (DINNER - 1);
  int bt = idx >> 9;
  int t  = bt & (SEQL - 1);
  float acc = cb[d];
  #pragma unroll
  for (int k = 0; k < 4; ++k) {
    int tt = t + k - 3;
    if (tt >= 0) acc += (float)xi[(size_t)(bt + k - 3) * DINNER + d] * cw[k * DINNER + d];
  }
  xc[idx] = (_Float16)(acc / (1.f + __expf(-acc)));
}

// z (only last token): z = h_last @ in_proj_w[:, 512:1024]
__global__ void zlast_kernel(const float* __restrict__ resid, const float* __restrict__ wi,
                             float* __restrict__ z) {
  int idx = blockIdx.x * blockDim.x + threadIdx.x;   // 64*512
  if (idx >= BATCH * DINNER) return;
  int j = idx & (DINNER - 1);
  int b = idx >> 9;
  const float* r = resid + b * DMODEL;
  float acc = 0.f;
  for (int k = 0; k < DMODEL; ++k) acc += r[k] * wi[(size_t)k * 1024 + DINNER + j];
  z[idx] = acc;
}

// C (only last token): C = xc_last @ x_proj_w[:, 80:144]
__global__ void clast_kernel(const _Float16* __restrict__ xc, const float* __restrict__ wx,
                             float* __restrict__ C) {
  int idx = blockIdx.x * blockDim.x + threadIdx.x;   // 64*64
  if (idx >= BATCH * NST) return;
  int n = idx & (NST - 1);
  int b = idx >> 6;
  const _Float16* xr = xc + ((size_t)b * SEQL + SEQL - 1) * DINNER;
  float acc = 0.f;
  for (int d = 0; d < DINNER; ++d) acc += (float)xr[d] * wx[(size_t)d * 144 + DBC + n];
  C[idx] = acc;
}

// ---------------------------------------------------------------------------
// Selective scan. 256 blocks = (batch, 128-channel block); 512 threads each =
// (128 channels x 4 state-quarters). 16 state + 16 A values per thread in
// registers; dt_proj + softplus fused per timestep via LDS weight slice.
// Writes the final state h_{T-1}.
// ---------------------------------------------------------------------------
__global__ void scan_kernel(const float* __restrict__ dtB, const _Float16* __restrict__ xc,
                            const float* __restrict__ dtw, const float* __restrict__ dtb,
                            const float* __restrict__ A_log, float* __restrict__ hstate) {
  const int b       = blockIdx.x >> 2;
  const int dblk    = blockIdx.x & 3;
  const int tid     = threadIdx.x;        // 512
  const int d_local = tid >> 2;           // 0..127
  const int d       = dblk * 128 + d_local;
  const int nq      = tid & 3;
  const int n0      = nq * 16;

  __shared__ float dtw_s[DTRANK * 128];
  __shared__ float bias_s[128];
  __shared__ float B_s[NST];
  __shared__ float dt_s[DTRANK];
  __shared__ float delta_s[128];

  for (int i = tid; i < DTRANK * 128; i += blockDim.x) {
    int r = i >> 7, dl = i & 127;
    dtw_s[i] = dtw[r * DINNER + dblk * 128 + dl];
  }
  if (tid < 128) bias_s[tid] = dtb[dblk * 128 + tid];

  float h[16], Ar[16];
  #pragma unroll
  for (int i = 0; i < 16; ++i) {
    h[i]  = 0.f;
    Ar[i] = -__expf(A_log[(size_t)d * NST + n0 + i]);
  }
  __syncthreads();

  const float*     dtBb = dtB + (size_t)b * SEQL * DBC;
  const _Float16*  xcb  = xc  + (size_t)b * SEQL * DINNER;

  for (int t = 0; t < SEQL; ++t) {
    if (tid < NST)    B_s[tid]  = dtBb[t * DBC + DTRANK + tid];
    if (tid < DTRANK) dt_s[tid] = dtBb[t * DBC + tid];
    __syncthreads();
    if (nq == 0) {
      float s = bias_s[d_local];
      #pragma unroll
      for (int r = 0; r < DTRANK; ++r) s += dt_s[r] * dtw_s[r * 128 + d_local];
      delta_s[d_local] = (s > 20.f) ? s : log1pf(__expf(s));   // softplus
    }
    __syncthreads();
    float delta = delta_s[d_local];
    float u     = (float)xcb[t * DINNER + d];
    float du    = delta * u;
    #pragma unroll
    for (int i = 0; i < 16; ++i)
      h[i] = h[i] * __expf(delta * Ar[i]) + du * B_s[n0 + i];
    __syncthreads();
  }
  float* hp = hstate + ((size_t)b * DINNER + d) * NST + n0;
  #pragma unroll
  for (int i = 0; i < 16; ++i) hp[i] = h[i];
}

// ---------------------------------------------------------------------------
// Fused tail (one block per batch): y = h.C + u*D, y *= silu(z),
// out = y @ out_proj + residual, LayerNorm, relu-MLP head -> scalar.
// ---------------------------------------------------------------------------
__global__ void final_kernel(const float* __restrict__ hstate, const float* __restrict__ Clast,
                             const _Float16* __restrict__ xc, const float* __restrict__ Dv,
                             const float* __restrict__ z, const float* __restrict__ out_w,
                             const float* __restrict__ resid, const float* __restrict__ ln_g,
                             const float* __restrict__ ln_b, const float* __restrict__ r1w,
                             const float* __restrict__ r1b, const float* __restrict__ r2w,
                             const float* __restrict__ r2b, float* __restrict__ out) {
  const int b   = blockIdx.x;
  const int tid = threadIdx.x;            // 256
  __shared__ float C_s[NST];
  __shared__ float y_s[DINNER];
  __shared__ float o_s[DMODEL];
  __shared__ float red[256];
  __shared__ float hr_s[64];
  __shared__ float stat[2];

  if (tid < NST) C_s[tid] = Clast[b * NST + tid];
  __syncthreads();

  for (int d = tid; d < DINNER; d += 256) {
    const float* hp = hstate + ((size_t)b * DINNER + d) * NST;
    float yd = 0.f;
    #pragma unroll 8
    for (int n = 0; n < NST; ++n) yd += hp[n] * C_s[n];
    float u  = (float)xc[((size_t)b * SEQL + SEQL - 1) * DINNER + d];
    yd += u * Dv[d];
    float zv = z[b * DINNER + d];
    yd *= zv / (1.f + __expf(-zv));      // * silu(z)
    y_s[d] = yd;
  }
  __syncthreads();

  float o = resid[b * DMODEL + tid];
  for (int d = 0; d < DINNER; ++d) o += y_s[d] * out_w[(size_t)d * DMODEL + tid];

  red[tid] = o; __syncthreads();
  for (int s = 128; s > 0; s >>= 1) { if (tid < s) red[tid] += red[tid + s]; __syncthreads(); }
  if (tid == 0) stat[0] = red[0] * (1.f / DMODEL);
  __syncthreads();
  red[tid] = o * o; __syncthreads();
  for (int s = 128; s > 0; s >>= 1) { if (tid < s) red[tid] += red[tid + s]; __syncthreads(); }
  if (tid == 0) stat[1] = red[0] * (1.f / DMODEL);
  __syncthreads();

  float mu  = stat[0];
  float var = stat[1] - mu * mu;
  o_s[tid] = (o - mu) * rsqrtf(var + 1e-5f) * ln_g[tid] + ln_b[tid];
  __syncthreads();

  if (tid < 64) {
    float hj = r1b[tid];
    for (int c = 0; c < DMODEL; ++c) hj += o_s[c] * r1w[(size_t)c * 64 + tid];
    hr_s[tid] = fmaxf(hj, 0.f);
  }
  __syncthreads();
  if (tid == 0) {
    float r = r2b[0];
    for (int j = 0; j < 64; ++j) r += hr_s[j] * r2w[j];
    out[b] = r;
  }
}

// ---------------------------------------------------------------------------
extern "C" void kernel_launch(void* const* d_in, const int* in_sizes, int n_in,
                              void* d_out, int out_size, void* d_ws, size_t ws_size,
                              hipStream_t stream) {
  const float* x         = (const float*)d_in[0];
  const float* embed_w   = (const float*)d_in[1];
  const float* embed_b   = (const float*)d_in[2];
  const float* in_proj_w = (const float*)d_in[3];
  const float* conv_w    = (const float*)d_in[4];
  const float* conv_b    = (const float*)d_in[5];
  const float* x_proj_w  = (const float*)d_in[6];
  const float* dt_proj_w = (const float*)d_in[7];
  const float* dt_proj_b = (const float*)d_in[8];
  const float* A_log     = (const float*)d_in[9];
  const float* Dv        = (const float*)d_in[10];
  const float* out_pw    = (const float*)d_in[11];
  const float* ln_g      = (const float*)d_in[12];
  const float* ln_b      = (const float*)d_in[13];
  const float* r1w       = (const float*)d_in[14];
  const float* r1b       = (const float*)d_in[15];
  const float* r2w       = (const float*)d_in[16];
  const float* r2b       = (const float*)d_in[17];
  float* out = (float*)d_out;

  char* ws = (char*)d_ws;
  size_t off = 0;
  auto carve = [&](size_t bytes) { void* p = ws + off; off = (off + bytes + 255) & ~(size_t)255; return p; };

  _Float16* h16   = (_Float16*)carve((size_t)ROWS * DMODEL * 2);   // 33.5 MB
  _Float16* xi16  = (_Float16*)carve((size_t)ROWS * DINNER * 2);   // 67 MB
  _Float16* xc16  = (_Float16*)carve((size_t)ROWS * DINNER * 2);   // 67 MB
  float*    dtB   = (float*)   carve((size_t)ROWS * DBC * 4);      // 21 MB
  float*    hst   = (float*)   carve((size_t)BATCH * DINNER * NST * 4); // 8.4 MB
  float*    resid = (float*)   carve((size_t)BATCH * DMODEL * 4);
  float*    zl    = (float*)   carve((size_t)BATCH * DINNER * 4);
  float*    Cl    = (float*)   carve((size_t)BATCH * NST * 4);
  _Float16* WpIn  = (_Float16*)carve((size_t)DMODEL * DINNER * 2); // packed in_proj[:, :512]
  _Float16* WpXp  = (_Float16*)carve((size_t)DINNER * DBC * 2);    // packed x_proj[:, :80]

  // 1) pack weights into WMMA B layout
  {
    int total = DMODEL * DINNER;               // K=256, 32 n-tiles
    pack_b_f16<<<(total + 255) / 256, 256, 0, stream>>>(in_proj_w, 1024, 0, DMODEL, WpIn, total);
  }
  {
    int total = DINNER * DBC;                  // K=512, 5 n-tiles
    pack_b_f16<<<(total + 255) / 256, 256, 0, stream>>>(x_proj_w, 144, 0, DINNER, WpXp, total);
  }

  // 2) embed (+ residual capture at t = SEQ-1)
  embed_kernel<<<(ROWS * DMODEL) / 256, 256, 0, stream>>>(x, embed_w, embed_b, h16, resid);

  // 3) in_proj xi-half: (65536x512) = h16 @ WpIn, K=256  -> WMMA
  {
    int waves = (ROWS / 16) * (DINNER / 64);   // NT=4 -> 64 cols/wave
    gemm_wmma<4, _Float16><<<waves / 4, 128, 0, stream>>>(h16, WpIn, xi16, ROWS, DINNER, DMODEL, DINNER);
  }

  // 4) depthwise conv + SiLU
  conv_silu<<<(ROWS * DINNER) / 256, 256, 0, stream>>>(xi16, conv_w, conv_b, xc16);

  // 5) x_proj dt+B: (65536x80) = xc16 @ WpXp, K=512 -> WMMA
  {
    int waves = (ROWS / 16);                   // NT=5 -> exactly 80 cols/wave
    gemm_wmma<5, float><<<waves / 4, 128, 0, stream>>>(xc16, WpXp, dtB, ROWS, DBC, DINNER, DBC);
  }

  // 6) last-token-only projections
  zlast_kernel<<<(BATCH * DINNER) / 256, 256, 0, stream>>>(resid, in_proj_w, zl);
  clast_kernel<<<(BATCH * NST) / 256, 256, 0, stream>>>(xc16, x_proj_w, Cl);

  // 7) selective scan -> final state
  scan_kernel<<<BATCH * 4, 512, 0, stream>>>(dtB, xc16, dt_proj_w, dt_proj_b, A_log, hst);

  // 8) fused tail
  final_kernel<<<BATCH, 256, 0, stream>>>(hst, Cl, xc16, Dv, zl, out_pw, resid,
                                          ln_g, ln_b, r1w, r1b, r2w, r2b, out);
}
